// SageNet_20117626814683
// MI455X (gfx1250) — compile-verified, matrix operand order checked by reading, not run on hardware
//
#include <hip/hip_runtime.h>
#include <math.h>

#define NEG_SLOPE 0.01f
#define L2EPS 1e-12f

typedef __attribute__((ext_vector_type(16))) _Float16 v16h;
typedef __attribute__((ext_vector_type(8)))  float    v8f;
typedef __attribute__((ext_vector_type(4)))  float    f4v;
typedef __attribute__((ext_vector_type(4)))  _Float16 h4;

// ---------------------------------------------------------------- zero fill
__global__ void sage_zero_kernel(float* __restrict__ p, long n) {
    long i = (long)blockIdx.x * blockDim.x + threadIdx.x;
    if (i < n) p[i] = 0.0f;
}

// ------------------------------------------------------- edge scatter-add
// grid = E blocks, block = d/4 threads. agg[dst] += h[src] via float4 loads
// + 4x global_atomic_add_f32. Feature buffers (~51MB) are L2-resident on
// MI455X's 192MB L2, so this phase is L2-atomic-bandwidth bound.
__global__ void sage_scatter_add_kernel(const float* __restrict__ h,
                                        const int* __restrict__ src,
                                        const int* __restrict__ dst,
                                        float* __restrict__ agg, int d) {
    int e = blockIdx.x;
    int t = threadIdx.x;                 // 0 .. d/4-1
    int s  = src[e];
    int dn = dst[e];
    f4v v = *(const f4v*)(h + (long)s * d + t * 4);
    float* base = agg + (long)dn * d + t * 4;
    atomicAdd(base + 0, v[0]);
    atomicAdd(base + 1, v[1]);
    atomicAdd(base + 2, v[2]);
    atomicAdd(base + 3, v[3]);
}

// --------------------------------------------------------- WMMA GEMM
// out[Nn, dout] = A[Nn, K] @ W[K, dout]   (f16 inputs, f32 accumulate)
// Block: 256 threads = 8 waves; block tile 64 rows x 32 cols; each wave
// computes one 16x16 tile via v_wmma_f32_16x16x32_f16; K fully unrolled.
template <int K>
__global__ void __launch_bounds__(256)
sage_gemm_wmma_kernel(const float* __restrict__ A, const float* __restrict__ W,
                      float* __restrict__ out, int Nn, int dout) {
    // 40-half row stride keeps the 8-half fragment reads 16B aligned
    __shared__ __align__(16) _Float16 sA[64][40];   // [row][k]
    __shared__ __align__(16) _Float16 sB[32][40];   // [col][k]  (W transposed)

    const int tid  = threadIdx.x;
    const int lane = tid & 31;
    const int wave = tid >> 5;
    const int wr   = wave >> 1;          // 0..3  (16-row sub-tile)
    const int wc   = wave & 1;           // 0..1  (16-col sub-tile)
    const int rowBase = blockIdx.x * 64;
    const int colBase = blockIdx.y * 32;

    // ---- fixed staging assignments (branchless; address clamped) ----
    // A: thread -> (row ar, 8-wide k chunk at ac). 64 rows x 4 chunks = 256.
    const int ar = tid >> 2;
    const int ac = (tid & 3) * 8;
    const int arow  = rowBase + ar;
    const int arowc = (arow < Nn) ? arow : (Nn - 1);   // clamp: load stays unconditional
    const float ascale = (arow < Nn) ? 1.0f : 0.0f;    // zero ragged rows w/o branching
    const float* aptr = A + (long)arowc * K + ac;

    // B: thread -> (k row bk, 4 adjacent cols at bc). 32 k x 8 groups = 256.
    const int bk = tid >> 3;
    const int bc = (tid & 7) * 4;
    const float* wptr = W + (long)bk * dout + colBase + bc;

    // fragment read pointers (CDNA5 16-bit A/B layout):
    // lanes 0-15: M(N)=lane, K = {0..7, 16..23}; lanes 16-31: K = {8..15, 24..31}
    const int m  = lane & 15;
    const int hi = lane >> 4;
    const _Float16* pa = &sA[wr * 16 + m][hi * 8];
    const _Float16* pb = &sB[wc * 16 + m][hi * 8];

    v8f acc = {};

    #pragma unroll
    for (int k0 = 0; k0 < K; k0 += 32) {
        // ---- stage A: two global_load_b128 -> two b64 LDS stores
        f4v a0 = *(const f4v*)(aptr + k0);
        f4v a1 = *(const f4v*)(aptr + k0 + 4);
        h4 ha0, ha1;
        #pragma unroll
        for (int i = 0; i < 4; ++i) {
            ha0[i] = (_Float16)(a0[i] * ascale);
            ha1[i] = (_Float16)(a1[i] * ascale);
        }
        *(h4*)&sA[ar][ac]     = ha0;
        *(h4*)&sA[ar][ac + 4] = ha1;

        // ---- stage B: one global_load_b128, transposed 4x ds_store_b16
        f4v w0 = *(const f4v*)(wptr + (long)k0 * dout);
        sB[bc + 0][bk] = (_Float16)w0[0];
        sB[bc + 1][bk] = (_Float16)w0[1];
        sB[bc + 2][bk] = (_Float16)w0[2];
        sB[bc + 3][bk] = (_Float16)w0[3];
        __syncthreads();

        // ---- fragments: 2x ds_load_b128 each
        v16h a, b;
        #pragma unroll
        for (int i = 0; i < 8; ++i) {
            a[i]     = pa[i];
            a[i + 8] = pa[i + 16];
            b[i]     = pb[i];
            b[i + 8] = pb[i + 16];
        }

        acc = __builtin_amdgcn_wmma_f32_16x16x32_f16(
                  /*neg_a=*/false, a, /*neg_b=*/false, b,
                  /*c_mod=*/(short)0, acc, /*reuse_a=*/false, /*reuse_b=*/false);
        __syncthreads();
    }

    // ---- store C/D fragment: VGPR r -> M = r + 8*(lane>=16), N = lane&15
    #pragma unroll
    for (int r = 0; r < 8; ++r) {
        int grow = rowBase + wr * 16 + hi * 8 + r;
        int gcol = colBase + wc * 16 + m;
        if (grow < Nn) out[(long)grow * dout + gcol] = acc[r];
    }
}

// -------------------------------------- bias + L2 normalize + leaky-relu
// grid = Nn blocks, block = d threads (d = 256 or 64)
__global__ void sage_norm_kernel(float* __restrict__ h, const float* __restrict__ bias,
                                 int d, int applyLrelu) {
    __shared__ float red[256];
    int r = blockIdx.x;
    int t = threadIdx.x;
    float v = h[(long)r * d + t] + bias[t];
    red[t] = v * v;
    __syncthreads();
    for (int s = blockDim.x >> 1; s > 0; s >>= 1) {
        if (t < s) red[t] += red[t + s];
        __syncthreads();
    }
    float nrm   = sqrtf(red[0]);
    float scale = 1.0f / fmaxf(nrm, L2EPS);
    v *= scale;
    if (applyLrelu) v = (v > 0.0f) ? v : NEG_SLOPE * v;
    h[(long)r * d + t] = v;
}

// --------------------------------------- gather first node of each graph
__global__ void sage_gather_kernel(const float* __restrict__ h, float* __restrict__ out,
                                   int stride, int dout, int total) {
    int i = blockIdx.x * blockDim.x + threadIdx.x;
    if (i < total) {
        int g = i / dout;
        int j = i - g * dout;
        out[i] = h[(long)(g * stride) * dout + j];
    }
}

extern "C" void kernel_launch(void* const* d_in, const int* in_sizes, int n_in,
                              void* d_out, int out_size, void* d_ws, size_t ws_size,
                              hipStream_t stream) {
    const float* x  = (const float*)d_in[0];
    const int*   ei = (const int*)d_in[1];
    // d_in[2] = batch (unused: graphs are contiguous blocks of N/G nodes)
    const float* W1 = (const float*)d_in[3];
    const float* b1 = (const float*)d_in[4];
    const float* W2 = (const float*)d_in[5];
    const float* b2 = (const float*)d_in[6];
    const float* W3 = (const float*)d_in[7];
    const float* b3 = (const float*)d_in[8];
    float* out = (float*)d_out;

    const int FIN = 128, HID = 256, FOUT = 64;
    const int Nn = in_sizes[0] / FIN;     // 50000
    const int E  = in_sizes[1] / 2;       // 800000
    const int G  = out_size / FOUT;       // 500
    const int stride = Nn / G;            // 100
    const int* src = ei;
    const int* dst = ei + E;

    float* agg = (float*)d_ws;                    // [Nn, 256] scratch
    float* h   = agg + (size_t)Nn * HID;          // [Nn, 256] activations

    const unsigned rowBlocks = (unsigned)((Nn + 63) / 64);

    // ---------------- layer 1: x (128) -> h (256), leaky-relu
    {
        long n = (long)Nn * FIN;
        sage_zero_kernel<<<(unsigned)((n + 255) / 256), 256, 0, stream>>>(agg, n);
        sage_scatter_add_kernel<<<E, FIN / 4, 0, stream>>>(x, src, dst, agg, FIN);
        dim3 grid(rowBlocks, HID / 32);
        sage_gemm_wmma_kernel<128><<<grid, 256, 0, stream>>>(agg, W1, h, Nn, HID);
        sage_norm_kernel<<<Nn, HID, 0, stream>>>(h, b1, HID, 1);
    }
    // ---------------- layer 2: h (256) -> h (256), leaky-relu
    {
        long n = (long)Nn * HID;
        sage_zero_kernel<<<(unsigned)((n + 255) / 256), 256, 0, stream>>>(agg, n);
        sage_scatter_add_kernel<<<E, HID / 4, 0, stream>>>(h, src, dst, agg, HID);
        dim3 grid(rowBlocks, HID / 32);
        sage_gemm_wmma_kernel<256><<<grid, 256, 0, stream>>>(agg, W2, h, Nn, HID);
        sage_norm_kernel<<<Nn, HID, 0, stream>>>(h, b2, HID, 1);
    }
    // ---------------- layer 3: h (256) -> h (64), no leaky-relu
    {
        long n = (long)Nn * HID;
        sage_zero_kernel<<<(unsigned)((n + 255) / 256), 256, 0, stream>>>(agg, n);
        sage_scatter_add_kernel<<<E, HID / 4, 0, stream>>>(h, src, dst, agg, HID);
        dim3 grid(rowBlocks, FOUT / 32);
        sage_gemm_wmma_kernel<256><<<grid, 256, 0, stream>>>(agg, W3, h, Nn, FOUT);
        sage_norm_kernel<<<Nn, FOUT, 0, stream>>>(h, b3, FOUT, 0);
    }

    sage_gather_kernel<<<(G * FOUT + 255) / 256, 256, 0, stream>>>(h, out, stride, FOUT, G * FOUT);
}
